// LegacySelfAttn1d_46394236731948
// MI455X (gfx1250) — matrix operand order repaired and checked by reading, not compiled, not run on hardware
//
#include <hip/hip_runtime.h>

// ---------------------------------------------------------------------------
// LegacySelfAttn1d on MI455X (gfx1250): bf16 WMMA pipeline
//   1) LayerNorm (f32 in) -> norm_bf16, (norm+pos)_bf16
//   2) f32->bf16 weight conversion
//   3) GEMM qk = posn @ Wqk^T + b  -> Q[B,H,N,D], K[B,H,N,D]   (WMMA bf16)
//   4) GEMM v  = norm @ Wv^T  + b  -> Vt[B,H,D,N] (transposed) (WMMA bf16)
//   5) Flash attention, 64 keys/iter (WMMA QK^T + WMMA P*V)
//   6) GEMM out = in + wv @ Wo^T + b (f32 out)                 (WMMA bf16)
// ---------------------------------------------------------------------------

typedef __attribute__((ext_vector_type(16))) __bf16 v16bf;
typedef __attribute__((ext_vector_type(8)))  __bf16 v8bf;
typedef __attribute__((ext_vector_type(4)))  __bf16 v4bf;
typedef __attribute__((ext_vector_type(8)))  float  v8f;

#define NB   4
#define NH   16
#define SEQ  1024
#define DH   64
#define FDIM 1024
#define MROWS (NB * SEQ)   // 4096
#define LDSROW 72          // padded P-tile row stride (bank-conflict-free b128)

__device__ __forceinline__ v8f wmma_bf16(v16bf a, v16bf b, v8f c) {
  return __builtin_amdgcn_wmma_f32_16x16x32_bf16(
      /*neg_a=*/false, a, /*neg_b=*/false, b,
      /*c_mod=*/(short)0, c, /*reuse_a=*/false, /*reuse_b=*/false);
}

// Load a 16-element bf16 fragment as two contiguous 16-byte chunks.
__device__ __forceinline__ v16bf load_frag(const __bf16* p0, const __bf16* p1) {
  v8bf lo = *reinterpret_cast<const v8bf*>(p0);
  v8bf hi = *reinterpret_cast<const v8bf*>(p1);
  v16bf r;
#pragma unroll
  for (int i = 0; i < 8; ++i) { r[i] = lo[i]; r[i + 8] = hi[i]; }
  return r;
}

// ---------------------------------------------------------------------------
// 1) LayerNorm: one block per (b,n) row of 1024 floats.
// ---------------------------------------------------------------------------
__global__ __launch_bounds__(256) void ln_kernel(
    const float* __restrict__ x, const float* __restrict__ pos,
    const float* __restrict__ g, const float* __restrict__ beta,
    __bf16* __restrict__ nrm, __bf16* __restrict__ pn) {
  __shared__ float ssum[256], ssq[256];
  const int row = blockIdx.x;
  const int t = threadIdx.x;
  const float* xr = x + (size_t)row * FDIM;
  float4 v = reinterpret_cast<const float4*>(xr)[t];
  float s = v.x + v.y + v.z + v.w;
  float q = v.x * v.x + v.y * v.y + v.z * v.z + v.w * v.w;
  ssum[t] = s; ssq[t] = q;
  __syncthreads();
  for (int o = 128; o > 0; o >>= 1) {
    if (t < o) { ssum[t] += ssum[t + o]; ssq[t] += ssq[t + o]; }
    __syncthreads();
  }
  const float mu = ssum[0] * (1.0f / FDIM);
  const float var = ssq[0] * (1.0f / FDIM) - mu * mu;
  const float rstd = rsqrtf(var + 1e-5f);

  float4 p4 = reinterpret_cast<const float4*>(pos + (size_t)row * FDIM)[t];
  float4 g4 = reinterpret_cast<const float4*>(g)[t];
  float4 b4 = reinterpret_cast<const float4*>(beta)[t];
  float n0 = (v.x - mu) * rstd * g4.x + b4.x;
  float n1 = (v.y - mu) * rstd * g4.y + b4.y;
  float n2 = (v.z - mu) * rstd * g4.z + b4.z;
  float n3 = (v.w - mu) * rstd * g4.w + b4.w;
  v4bf nv = { (__bf16)n0, (__bf16)n1, (__bf16)n2, (__bf16)n3 };
  v4bf pv = { (__bf16)(n0 + p4.x), (__bf16)(n1 + p4.y),
              (__bf16)(n2 + p4.z), (__bf16)(n3 + p4.w) };
  *reinterpret_cast<v4bf*>(nrm + (size_t)row * FDIM + t * 4) = nv;
  *reinterpret_cast<v4bf*>(pn  + (size_t)row * FDIM + t * 4) = pv;
}

// ---------------------------------------------------------------------------
// 2) f32 -> bf16 conversion (weights)
// ---------------------------------------------------------------------------
__global__ __launch_bounds__(256) void cvt_bf16_kernel(
    const float* __restrict__ in, __bf16* __restrict__ out, int n) {
  int i = blockIdx.x * blockDim.x + threadIdx.x;
  const int stride = gridDim.x * blockDim.x;
  for (; i < n; i += stride) out[i] = (__bf16)in[i];
}

// ---------------------------------------------------------------------------
// 3/4/6) Register-blocked WMMA GEMM: C[M,Nout] = A[M,1024] @ W[Nout,1024]^T + b
//   256 threads = 8 waves, each wave owns a 64x64 tile (block = 128x256).
//   Double-buffered K loop (64-deep per step, two 32-deep fragment sets).
//   MODE 0: split-store qk -> Q[B,H,N,D] / K[B,H,N,D]
//   MODE 1: store v transposed -> Vt[B,H,D,N]
//   MODE 2: f32 out = resid + C
// ---------------------------------------------------------------------------
__device__ __forceinline__ void load_tiles(
    const __bf16* __restrict__ A, const __bf16* __restrict__ W,
    int m0, int n0, int lr, int lg, int k0,
    v16bf af[4], v16bf bw[4]) {
#pragma unroll
  for (int mi = 0; mi < 4; ++mi) {
    // A fragment (16x32): lane=row, K chunks at lg*8 and 16+lg*8
    const __bf16* p = A + (size_t)(m0 + mi * 16 + lr) * FDIM + k0 + lg * 8;
    af[mi] = load_frag(p, p + 16);
  }
#pragma unroll
  for (int ni = 0; ni < 4; ++ni) {
    // B fragment (32x16): lane=col, K chunks at lg*8 and 16+lg*8
    const __bf16* p = W + (size_t)(n0 + ni * 16 + lr) * FDIM + k0 + lg * 8;
    bw[ni] = load_frag(p, p + 16);
  }
}

__device__ __forceinline__ void mma_tiles(const v16bf af[4], const v16bf bw[4],
                                          v8f acc[4][4]) {
#pragma unroll
  for (int mi = 0; mi < 4; ++mi)
#pragma unroll
    for (int ni = 0; ni < 4; ++ni)
      acc[mi][ni] = wmma_bf16(af[mi], bw[ni], acc[mi][ni]);
}

template <int MODE>
__global__ __launch_bounds__(256) void gemm_kernel(
    const __bf16* __restrict__ A, const __bf16* __restrict__ W,
    const float* __restrict__ bias,
    __bf16* __restrict__ out0, __bf16* __restrict__ out1,
    float* __restrict__ fout, const float* __restrict__ resid) {
  const int wave = threadIdx.x >> 5;
  const int l = threadIdx.x & 31;
  const int lr = l & 15, lg = l >> 4;
  const int m0 = blockIdx.y * 128 + (wave >> 2) * 64;
  const int n0 = blockIdx.x * 256 + (wave & 3) * 64;

  v8f acc[4][4] = {};
  v16bf af0[4], bw0[4], af1[4], bw1[4];

  load_tiles(A, W, m0, n0, lr, lg, 0, af0, bw0);
  for (int k0 = 0; k0 < FDIM; k0 += 64) {
    load_tiles(A, W, m0, n0, lr, lg, k0 + 32, af1, bw1);
    mma_tiles(af0, bw0, acc);
    if (k0 + 64 < FDIM)
      load_tiles(A, W, m0, n0, lr, lg, k0 + 64, af0, bw0);
    mma_tiles(af1, bw1, acc);
  }

#pragma unroll
  for (int mi = 0; mi < 4; ++mi)
#pragma unroll
    for (int ni = 0; ni < 4; ++ni) {
      const int col = n0 + ni * 16 + lr;
      const float bv = bias[col];
#pragma unroll
      for (int r = 0; r < 8; ++r) {
        const int grow = m0 + mi * 16 + r + 8 * lg;  // C: row = vgpr + 8*lg
        const float val = acc[mi][ni][r] + bv;
        if (MODE == 0) {
          const int b = grow >> 10, n = grow & 1023;
          if (col < FDIM) {
            const int hh = col >> 6, d = col & 63;
            out0[(((size_t)b * NH + hh) * SEQ + n) * DH + d] = (__bf16)val;
          } else {
            const int c2 = col - FDIM, hh = c2 >> 6, d = c2 & 63;
            out1[(((size_t)b * NH + hh) * SEQ + n) * DH + d] = (__bf16)val;
          }
        } else if (MODE == 1) {
          const int b = grow >> 10, n = grow & 1023;
          const int hh = col >> 6, d = col & 63;
          out0[(((size_t)b * NH + hh) * DH + d) * SEQ + n] = (__bf16)val;
        } else {
          const size_t idx = (size_t)grow * FDIM + col;
          fout[idx] = resid[idx] + val;
        }
      }
    }
}

// ---------------------------------------------------------------------------
// 5) Flash attention: 128 threads = 4 waves; each wave owns 16 query rows of
//    one (b,h). Streams 64 keys/iteration: 8 WMMAs QK^T + online softmax +
//    8 WMMAs P*V. P is reshaped C-layout -> A-layout through a private,
//    pad-72 LDS tile (no barriers needed: same-wave DS ops are in-order).
// ---------------------------------------------------------------------------
__global__ __launch_bounds__(128) void attn_kernel(
    const __bf16* __restrict__ Q, const __bf16* __restrict__ K,
    const __bf16* __restrict__ Vt, __bf16* __restrict__ wv) {
  __shared__ __bf16 plds[4][16 * LDSROW];
  const int wave = threadIdx.x >> 5;
  const int l = threadIdx.x & 31;
  const int lr = l & 15, lg = l >> 4, koff = lg * 8;
  const int bh = blockIdx.x >> 4;                    // b*16 + h
  const int m0 = (blockIdx.x & 15) * 64 + wave * 16; // query tile base
  const int b = bh >> 4, h = bh & 15;

  const __bf16* Qh = Q  + (size_t)bh * SEQ * DH;
  const __bf16* Kh = K  + (size_t)bh * SEQ * DH;
  const __bf16* Vh = Vt + (size_t)bh * DH * SEQ;

  // Preload Q A-fragments covering d = 0..63 (two 16x32 fragments)
  const __bf16* qp = Qh + (size_t)(m0 + lr) * DH;
  const v16bf aq0 = load_frag(qp + koff,      qp + 16 + koff);
  const v16bf aq1 = load_frag(qp + 32 + koff, qp + 48 + koff);

  v8f o[4] = {};
  float mrun[8], lrun[8];
#pragma unroll
  for (int r = 0; r < 8; ++r) { mrun[r] = -1e30f; lrun[r] = 0.0f; }
  const float scale = 0.125f;  // 1/sqrt(64)
  __bf16* pl = plds[wave];

  for (int j = 0; j < SEQ; j += 64) {
    // S = Q * K^T for a 16x64 score block (four 16x16 C tiles)
    v8f s[4];
#pragma unroll
    for (int si = 0; si < 4; ++si) {
      const __bf16* kp = Kh + (size_t)(j + si * 16 + lr) * DH + koff;
      v16bf bk0 = load_frag(kp,      kp + 16);      // d = 0..31
      v16bf bk1 = load_frag(kp + 32, kp + 48);      // d = 32..63
      v8f c = {};
      c = wmma_bf16(aq0, bk0, c);
      c = wmma_bf16(aq1, bk1, c);
      s[si] = c;
    }
    if (j + 64 < SEQ)
      __builtin_prefetch(Kh + (size_t)(j + 64 + lr) * DH);  // global_prefetch

    // Online softmax update (rows live across 16-lane half-waves)
    float alpha[8];
#pragma unroll
    for (int r = 0; r < 8; ++r) {
      const float x0 = s[0][r] * scale, x1 = s[1][r] * scale;
      const float x2 = s[2][r] * scale, x3 = s[3][r] * scale;
      float bm = fmaxf(fmaxf(x0, x1), fmaxf(x2, x3));
#pragma unroll
      for (int off = 1; off < 16; off <<= 1) bm = fmaxf(bm, __shfl_xor(bm, off, 16));
      const float mnew = fmaxf(mrun[r], bm);
      const float a_ = __expf(mrun[r] - mnew);
      const float p0 = __expf(x0 - mnew), p1 = __expf(x1 - mnew);
      const float p2 = __expf(x2 - mnew), p3 = __expf(x3 - mnew);
      float rs = (p0 + p1) + (p2 + p3);
#pragma unroll
      for (int off = 1; off < 16; off <<= 1) rs += __shfl_xor(rs, off, 16);
      lrun[r] = lrun[r] * a_ + rs;
      mrun[r] = mnew;
      alpha[r] = a_;
      s[0][r] = p0; s[1][r] = p1; s[2][r] = p2; s[3][r] = p3;
    }
#pragma unroll
    for (int t = 0; t < 4; ++t)
#pragma unroll
      for (int r = 0; r < 8; ++r) o[t][r] *= alpha[r];

    // Reshape P (C-layout) -> A-fragment layout through padded LDS tile.
    // Same-wave LDS ops are in-order (DScnt), wave-private region: no barrier.
#pragma unroll
    for (int r = 0; r < 8; ++r) {
      const int row = r + 8 * lg;
#pragma unroll
      for (int si = 0; si < 4; ++si)
        pl[row * LDSROW + si * 16 + lr] = (__bf16)s[si][r];
    }
    const __bf16* app = pl + lr * LDSROW + koff;
    const v16bf ap0 = load_frag(app,      app + 16);  // keys j..j+31
    const v16bf ap1 = load_frag(app + 32, app + 48);  // keys j+32..j+63

    // O += P * V  (V transposed: contiguous key runs per d-column)
#pragma unroll
    for (int t = 0; t < 4; ++t) {
      const __bf16* vp = Vh + (size_t)(t * 16 + lr) * SEQ + j + koff;
      v16bf bv0 = load_frag(vp,      vp + 16);
      v16bf bv1 = load_frag(vp + 32, vp + 48);
      o[t] = wmma_bf16(ap0, bv0, o[t]);
      o[t] = wmma_bf16(ap1, bv1, o[t]);
    }
  }

  // Normalize (v_rcp_f32 is plenty for a softmax normalizer) and write
  // wv[B,N,F] (F = h*64 + d) for the output projection.
  float inv[8];
#pragma unroll
  for (int r = 0; r < 8; ++r) inv[r] = __builtin_amdgcn_rcpf(lrun[r]);
#pragma unroll
  for (int t = 0; t < 4; ++t)
#pragma unroll
    for (int r = 0; r < 8; ++r) {
      const int grow = m0 + r + 8 * lg;
      const int d = t * 16 + lr;
      wv[((size_t)b * SEQ + grow) * FDIM + h * DH + d] = (__bf16)(o[t][r] * inv[r]);
    }
}

// ---------------------------------------------------------------------------
// Host launcher
// ---------------------------------------------------------------------------
extern "C" void kernel_launch(void* const* d_in, const int* in_sizes, int n_in,
                              void* d_out, int out_size, void* d_ws, size_t ws_size,
                              hipStream_t stream) {
  const float* in_feats  = (const float*)d_in[0];
  const float* pos_feats = (const float*)d_in[1];
  const float* ln_g = (const float*)d_in[2];
  const float* ln_b = (const float*)d_in[3];
  const float* w_qk = (const float*)d_in[4];
  const float* b_qk = (const float*)d_in[5];
  const float* w_v  = (const float*)d_in[6];
  const float* b_v  = (const float*)d_in[7];
  const float* w_o  = (const float*)d_in[8];
  const float* b_o  = (const float*)d_in[9];
  float* out = (float*)d_out;

  char* w = (char*)d_ws;
  __bf16* norm_bf = (__bf16*)(w + 0);          //  8 MiB [B,N,F]
  __bf16* posn_bf = (__bf16*)(w + 8388608);    //  8 MiB [B,N,F]
  __bf16* wqk_bf  = (__bf16*)(w + 16777216);   //  4 MiB [2F,F]
  __bf16* wv_bf   = (__bf16*)(w + 20971520);   //  2 MiB [F,F]
  __bf16* wo_bf   = (__bf16*)(w + 23068672);   //  2 MiB [F,F]
  __bf16* Qb      = (__bf16*)(w + 25165824);   //  8 MiB [B,H,N,D]
  __bf16* Kb      = (__bf16*)(w + 33554432);   //  8 MiB [B,H,N,D]
  __bf16* Vtb     = (__bf16*)(w + 41943040);   //  8 MiB [B,H,D,N]
  __bf16* wvb     = (__bf16*)(w + 50331648);   //  8 MiB [B,N,F]

  // 1) LayerNorm
  ln_kernel<<<MROWS, 256, 0, stream>>>(in_feats, pos_feats, ln_g, ln_b,
                                       norm_bf, posn_bf);
  // 2) weight conversion
  cvt_bf16_kernel<<<1024, 256, 0, stream>>>(w_qk, wqk_bf, 2 * FDIM * FDIM);
  cvt_bf16_kernel<<<1024, 256, 0, stream>>>(w_v,  wv_bf,  FDIM * FDIM);
  cvt_bf16_kernel<<<1024, 256, 0, stream>>>(w_o,  wo_bf,  FDIM * FDIM);
  // 3) QK projection (Nout = 2048)
  gemm_kernel<0><<<dim3(8, 32), 256, 0, stream>>>(posn_bf, wqk_bf, b_qk,
                                                  Qb, Kb, nullptr, nullptr);
  // 4) V projection, stored transposed (Nout = 1024)
  gemm_kernel<1><<<dim3(4, 32), 256, 0, stream>>>(norm_bf, wv_bf, b_v,
                                                  Vtb, nullptr, nullptr, nullptr);
  // 5) attention
  attn_kernel<<<NB * NH * (SEQ / 64), 128, 0, stream>>>(Qb, Kb, Vtb, wvb);
  // 6) output projection + residual (f32 out)
  gemm_kernel<2><<<dim3(4, 32), 256, 0, stream>>>(wvb, wo_bf, b_o,
                                                  nullptr, nullptr, out, in_feats);
}
